// SynthesisBlock_62285615727388
// MI455X (gfx1250) — matrix operand order repaired and checked by reading, not compiled, not run on hardware
//
#include <hip/hip_runtime.h>
#include <hip/hip_bf16.h>
#include <math.h>

// ---------------- problem constants ----------------
#define Bn 16
#define Cc 64
#define RES 256
#define NPIX (RES*RES)          // 65536
#define WDIM 512
#define GROUPS 32
#define CH_PER_G (Cc/GROUPS)    // 2
#define SQRT2F 1.41421356237309504880f

typedef __attribute__((ext_vector_type(16))) __bf16 v16bf;
typedef __attribute__((ext_vector_type(8)))  __bf16 v8bf;
typedef __attribute__((ext_vector_type(8)))  float  v8f;

// ---------------- workspace layout (bytes) ----------------
// s        : 16*192*4            = 12288
// styles   : 16*64*4             = 4096
// stats    : 512*2*4             = 4096   (mean, rstd per (b,group))
// Bw       : 16*9*2*4*32*8*4     = 1179648  (swizzled bf16 weights, stored as dwords)
// xn bf16  : 16*65536*64*2       = 134217728
#define OFF_S       0
#define OFF_STYLES  12288
#define OFF_STATS   16384
#define OFF_BW      20480
#define OFF_XN      1310720
#define BW_DW_PER_B 18432   // 9*2*4*32*8 dwords per sample

// ================= kernel 1: s = w @ affine_w^T / sqrt(512) + affine_b =================
__global__ void k_affine(const float* __restrict__ w, const float* __restrict__ aw,
                         const float* __restrict__ ab, float* __restrict__ s_out) {
  int tid = blockIdx.x * blockDim.x + threadIdx.x;          // 16*192 = 3072
  if (tid >= Bn * 3 * Cc) return;
  int b = tid / (3 * Cc), j = tid % (3 * Cc);
  const float* wr = w + b * WDIM;
  const float* ar = aw + j * WDIM;
  float acc = 0.f;
  #pragma unroll 4
  for (int k = 0; k < WDIM; ++k) acc += wr[k] * ar[k];
  s_out[tid] = acc * 0.044194173824159216f /* 1/sqrt(512) */ + ab[j];
}

// ================= kernel 2: styles = m1*m2 + m3 =================
__global__ void k_styles(const float* __restrict__ s, float* __restrict__ styles) {
  int tid = blockIdx.x * blockDim.x + threadIdx.x;          // 16*64 = 1024
  if (tid >= Bn * Cc) return;
  int b = tid / Cc, c = tid % Cc;
  const float* sr = s + b * 3 * Cc;
  styles[tid] = sr[c] * sr[Cc + c] + sr[2 * Cc + c];
}

// ================= kernel 3: groupnorm statistics =================
__global__ void k_gnstats(const float* __restrict__ x, float* __restrict__ stats) {
  __shared__ float ssum[256], ssq[256];
  int blk = blockIdx.x;                 // 512 = b*32+g
  int b = blk / GROUPS, g = blk % GROUPS;
  int t = threadIdx.x;
  const float4* base = (const float4*)(x + ((size_t)(b * Cc + g * CH_PER_G)) * NPIX);
  const int n4 = CH_PER_G * NPIX / 4;   // 32768 float4
  float sum = 0.f, sq = 0.f;
  for (int i = t; i < n4; i += 256) {
    float4 v = base[i];
    sum += v.x + v.y + v.z + v.w;
    sq  += v.x * v.x + v.y * v.y + v.z * v.z + v.w * v.w;
  }
  ssum[t] = sum; ssq[t] = sq;
  __syncthreads();
  for (int off = 128; off > 0; off >>= 1) {
    if (t < off) { ssum[t] += ssum[t + off]; ssq[t] += ssq[t + off]; }
    __syncthreads();
  }
  if (t == 0) {
    const float inv_n = 1.f / (float)(CH_PER_G * NPIX);
    float mu = ssum[0] * inv_n;
    float var = ssq[0] * inv_n - mu * mu;
    stats[blk * 2 + 0] = mu;
    stats[blk * 2 + 1] = rsqrtf(var + 1e-5f);
  }
}

// ================= kernel 4: xn = GN(x) -> bf16 NHWC (LDS transpose) =================
__global__ void k_xn_bf16(const float* __restrict__ x, const float* __restrict__ stats,
                          const float* __restrict__ gn_w, const float* __restrict__ gn_b,
                          __bf16* __restrict__ xnb) {
  __shared__ __bf16 tile[64][64];       // [pix_local][channel], 8KB
  int b = blockIdx.x >> 10;             // 1024 tiles of 64 pixels per sample
  int pbase = (blockIdx.x & 1023) << 6;
  int t = threadIdx.x;
  int c = t >> 2, q = t & 3;
  int g = c >> 1;
  float mu = stats[(b * GROUPS + g) * 2 + 0];
  float rstd = stats[(b * GROUPS + g) * 2 + 1];
  float scale = rstd * gn_w[c];
  float shift = gn_b[c] - mu * scale;
  const float4* src = (const float4*)(x + ((size_t)(b * Cc + c)) * NPIX + pbase + q * 16);
  #pragma unroll
  for (int j = 0; j < 4; ++j) {
    float4 v = src[j];
    int p = q * 16 + j * 4;
    tile[p + 0][c] = (__bf16)(v.x * scale + shift);
    tile[p + 1][c] = (__bf16)(v.y * scale + shift);
    tile[p + 2][c] = (__bf16)(v.z * scale + shift);
    tile[p + 3][c] = (__bf16)(v.w * scale + shift);
  }
  __syncthreads();
  int pix = t >> 2;                     // 64 pixels, 4 threads each write 16 ch (32B)
  uint4 u0 = *(const uint4*)&tile[pix][q * 16];
  uint4 u1 = *(const uint4*)&tile[pix][q * 16 + 8];
  uint4* dst = (uint4*)(xnb + ((size_t)(b * NPIX + pbase + pix)) * Cc + q * 16);
  dst[0] = u0; dst[1] = u1;
}

// ================= kernel 5: modulate+demod weights -> swizzled bf16 B-operands ========
// Layout per sample: [tap(9)][kstep(2)][ntile(4)][lane(32)][vgpr(8)] dwords,
// matching V_WMMA 16-bit B 32x16: lane<16 -> K=2v,2v+1 (v<4), 16+2(v-4) (v>=4) / N=lane ; lane>=16 -> same K+8 / N=lane-16
__global__ void k_weights(const float* __restrict__ conv_w, const float* __restrict__ styles,
                          unsigned int* __restrict__ Bw) {
  __shared__ float sty[Cc], dco[Cc];
  int b = blockIdx.x, t = threadIdx.x;
  if (t < Cc) sty[t] = styles[b * Cc + t];
  __syncthreads();
  if (t < Cc) {
    int o = t;
    float acc = 0.f;
    for (int i = 0; i < Cc; ++i) {
      float st = sty[i];
      const float* wp = conv_w + (o * Cc + i) * 9;
      #pragma unroll
      for (int k = 0; k < 9; ++k) { float v = wp[k] * st; acc += v * v; }
    }
    dco[o] = rsqrtf(acc + 1e-8f);
  }
  __syncthreads();
  unsigned int* out = Bw + (size_t)b * BW_DW_PER_B;
  for (int d = t; d < BW_DW_PER_B; d += 256) {
    int v     = d & 7;
    int lane  = (d >> 3) & 31;
    int ntile = (d >> 8) & 3;
    int ks    = (d >> 10) & 1;
    int tap   = d >> 11;
    int k0 = ks * 32 + ((lane >> 4) << 3) + 2 * v + ((v >> 2) << 3);
    int o  = ntile * 16 + (lane & 15);
    float sc = dco[o];
    float f0 = conv_w[(o * Cc + k0) * 9 + tap]     * sty[k0]     * sc;
    float f1 = conv_w[(o * Cc + k0 + 1) * 9 + tap] * sty[k0 + 1] * sc;
    __bf16 h0 = (__bf16)f0, h1 = (__bf16)f1;
    unsigned int lo = *(unsigned short*)&h0;
    unsigned int hi = *(unsigned short*)&h1;
    out[d] = lo | (hi << 16);
  }
}

// -------- A-fragment load: 16 pixels x 32 channels, zero-filled at SAME-pad edges -----
__device__ __forceinline__ v16bf load_a(const __bf16* pix, int ks, bool valid) {
  v16bf a;
  if (valid) {
    v8bf lo = *(const v8bf*)(pix + ks * 32);
    v8bf hi = *(const v8bf*)(pix + ks * 32 + 16);
    #pragma unroll
    for (int i = 0; i < 8; ++i) { a[i] = lo[i]; a[8 + i] = hi[i]; }
  } else {
    #pragma unroll
    for (int i = 0; i < 16; ++i) a[i] = (__bf16)0.0f;
  }
  return a;
}

// ================= kernel 6: implicit-GEMM conv (WMMA bf16) + fused epilogue ===========
// Each wave: 2 adjacent M-tiles (32 pixels) x 1 N-tile (16 out ch); B fragments are
// register-reused across the two back-to-back WMMAs. 36 WMMA / wave.
__global__ void __launch_bounds__(256)
k_conv_wmma(const __bf16* __restrict__ xnb, const unsigned int* __restrict__ Bw,
            const float* __restrict__ x, const float* __restrict__ stats,
            const float* __restrict__ gn_w, const float* __restrict__ gn_b,
            const float* __restrict__ conv_b, const float* __restrict__ noise_const,
            const float* __restrict__ noise_strength, const float* __restrict__ gamma,
            float* __restrict__ out) {
  // 1024 blocks per sample * 8 waves = 8192 wave-tiles = 2048 Mpairs x 4 Ntiles
  int b    = blockIdx.x >> 10;
  int rem  = blockIdx.x & 1023;
  int wave = threadIdx.x >> 5;
  int lane = threadIdx.x & 31;
  int wt    = rem * 8 + wave;
  int ntile = wt & 3;
  int mpair = wt >> 2;                 // 0..2047 ; 32 pixels each
  int h  = mpair >> 3;
  int w0 = (mpair & 7) << 5;           // pixels [w0, w0+32)

  int lhalf = lane >> 4;               // 0 or 1
  int mlane = lane & 15;               // pixel index within tile for A; N index for B/C
  v8f acc0 = {}, acc1 = {};

  const __bf16* xb = xnb + (size_t)b * NPIX * Cc;
  const unsigned int* bw = Bw + (size_t)b * BW_DW_PER_B + (size_t)ntile * 256 + (size_t)lane * 8;

  #pragma unroll
  for (int tap = 0; tap < 9; ++tap) {
    int dh = tap / 3 - 1, dw = tap % 3 - 1;
    int hh = h + dh;
    bool hok = (hh >= 0) & (hh < RES);
    int ws0 = w0 + mlane + dw;
    int ws1 = ws0 + 16;
    bool v0 = hok & (ws0 >= 0) & (ws0 < RES);
    bool v1 = hok & (ws1 < RES);       // ws1 >= 16+dw >= 15 >= 0 always
    const __bf16* pix0 = xb + ((size_t)(hh * RES + ws0)) * Cc + lhalf * 8;
    const __bf16* pix1 = pix0 + (size_t)16 * Cc;
    #pragma unroll
    for (int ks = 0; ks < 2; ++ks) {
      v16bf a0 = load_a(pix0, ks, v0);
      v16bf a1 = load_a(pix1, ks, v1);
      v16bf bm = *(const v16bf*)(bw + ((size_t)tap * 2 + ks) * 1024);
      acc0 = __builtin_amdgcn_wmma_f32_16x16x32_bf16(
          false, a0, false, bm, (short)0, acc0, false, false);
      acc1 = __builtin_amdgcn_wmma_f32_16x16x32_bf16(
          false, a1, false, bm, (short)0, acc1, false, false);
    }
  }

  // ---- fused epilogue (vectorized float4 loads/stores: 8 consecutive w per lane) ----
  int o = ntile * 16 + mlane;
  float cb  = conv_b[o];
  float gam = gamma[o];
  float ns  = noise_strength[0];
  int g = o >> 1;
  float mu   = stats[(b * GROUPS + g) * 2 + 0];
  float rstd = stats[(b * GROUPS + g) * 2 + 1];
  float scale = rstd * gn_w[o];
  float shift = gn_b[o] - mu * scale;
  const float* xrow = x + ((size_t)(b * Cc + o)) * NPIX + h * RES;
  float* orow = out + ((size_t)(b * Cc + o)) * NPIX + h * RES;
  const float* nrow = noise_const + h * RES;

  #pragma unroll
  for (int t = 0; t < 2; ++t) {
    const v8f acc = t ? acc1 : acc0;
    int wb = w0 + t * 16 + (lhalf << 3);     // 8 consecutive pixels start here
    const float4* n4 = (const float4*)(nrow + wb);
    const float4* x4 = (const float4*)(xrow + wb);
    float4*       o4 = (float4*)(orow + wb);
    #pragma unroll
    for (int half = 0; half < 2; ++half) {
      float4 nv = n4[half];
      float4 xv = x4[half];
      float4 ov;
      float nin[4] = {nv.x, nv.y, nv.z, nv.w};
      float xin[4] = {xv.x, xv.y, xv.z, xv.w};
      float res[4];
      #pragma unroll
      for (int j = 0; j < 4; ++j) {
        float y = acc[half * 4 + j];
        y += nin[j] * ns;
        y += cb;
        y = (fmaxf(y, 0.f) + 0.2f * fminf(y, 0.f)) * SQRT2F;   // lrelu(0.2)*sqrt(2)
        float xnv = xin[j] * scale + shift;                     // residual: normed x, f32
        res[j] = (gam * y + xnv) * SQRT2F;
      }
      ov.x = res[0]; ov.y = res[1]; ov.z = res[2]; ov.w = res[3];
      o4[half] = ov;
    }
  }
}

// ================= launch =================
extern "C" void kernel_launch(void* const* d_in, const int* in_sizes, int n_in,
                              void* d_out, int out_size, void* d_ws, size_t ws_size,
                              hipStream_t stream) {
  const float* x        = (const float*)d_in[0];
  const float* w        = (const float*)d_in[1];
  const float* aw       = (const float*)d_in[2];
  const float* ab       = (const float*)d_in[3];
  const float* conv_w   = (const float*)d_in[4];
  const float* conv_b   = (const float*)d_in[5];
  const float* gn_w     = (const float*)d_in[6];
  const float* gn_b     = (const float*)d_in[7];
  const float* noise_c  = (const float*)d_in[8];
  const float* noise_s  = (const float*)d_in[9];
  const float* gamma    = (const float*)d_in[10];
  float* out = (float*)d_out;

  char* ws = (char*)d_ws;
  float*        s_buf   = (float*)(ws + OFF_S);
  float*        styles  = (float*)(ws + OFF_STYLES);
  float*        stats   = (float*)(ws + OFF_STATS);
  unsigned int* Bw      = (unsigned int*)(ws + OFF_BW);
  __bf16*       xnb     = (__bf16*)(ws + OFF_XN);

  k_affine <<<(Bn * 3 * Cc + 255) / 256, 256, 0, stream>>>(w, aw, ab, s_buf);
  k_styles <<<(Bn * Cc + 255) / 256,      256, 0, stream>>>(s_buf, styles);
  k_gnstats<<<Bn * GROUPS,                256, 0, stream>>>(x, stats);
  k_xn_bf16<<<Bn * (NPIX / 64),           256, 0, stream>>>(x, stats, gn_w, gn_b, xnb);
  k_weights<<<Bn,                         256, 0, stream>>>(conv_w, styles, Bw);
  k_conv_wmma<<<Bn * 1024,                256, 0, stream>>>(
      xnb, Bw, x, stats, gn_w, gn_b, conv_b, noise_c, noise_s, gamma, out);
}
